// EncoderRNN_26946624815385
// MI455X (gfx1250) — matrix operand order repaired and checked by reading, not compiled, not run on hardware
//
#include <hip/hip_runtime.h>
#include <hip/hip_bf16.h>

// ---------------------------------------------------------------------------
// GRU (EncoderRNN): SEQ=8192, IN=1024, HID=1024
//   Phase 1: gx = inp @ W_ih^T + b_ih   (bf16 WMMA GEMM, f32 accumulate,
//            async global->LDS tile staging via GLOBAL_LOAD_ASYNC_TO_LDS_B128)
//   Phase 2: sequential scan, persistent kernel + device-wide barrier (f32)
// ---------------------------------------------------------------------------

typedef __bf16 bf16_t;
typedef __attribute__((ext_vector_type(16))) __bf16 v16bf;
typedef __attribute__((ext_vector_type(8)))  __bf16 v8bf;
typedef __attribute__((ext_vector_type(8)))  float  v8f;

static constexpr int SEQ = 8192;
static constexpr int IN  = 1024;
static constexpr int HID = 1024;
static constexpr int NO  = 3 * HID;   // 3072 gate rows

// ---------------- CDNA5 async global->LDS copy (ASYNCcnt path) --------------
// INST_OFFSET applies to BOTH the LDS and global addresses (ISA §10 async),
// so one LDS-addr VGPR + one global-addr VGPR pair covers 32 B in two b128s.
// Generic pointers to __shared__ carry the LDS byte offset in addr[31:0].
__device__ __forceinline__ void async_copy_32B(const void* gptr, void* lptr) {
    unsigned lds = (unsigned)(uintptr_t)lptr;
    asm volatile(
        "global_load_async_to_lds_b128 %0, %1, off\n\t"
        "global_load_async_to_lds_b128 %0, %1, off offset:16"
        :: "v"(lds), "v"(gptr) : "memory");
}
__device__ __forceinline__ void wait_async0() {
    asm volatile("s_wait_asynccnt 0" ::: "memory");
}

// ---------------------------- f32 -> bf16 cast ------------------------------
__global__ __launch_bounds__(256)
void cvt_bf16(const float* __restrict__ src, bf16_t* __restrict__ dst, int n) {
    int i = blockIdx.x * blockDim.x + threadIdx.x;
    int stride = gridDim.x * blockDim.x;
    for (; i < n; i += stride) dst[i] = (bf16_t)src[i];
}

// ------------------------- Phase 1: WMMA GEMM -------------------------------
// C[M=8192, N=3072] = A[M, K=1024] * B[N, K]^T + b_ih
// Block tile 128x128, K-step 32, 8 waves (4 along M x 2 along N),
// each wave owns a 32x64 sub-tile = 2x4 wmma 16x16 accumulators.
static constexpr int BM  = 128;
static constexpr int BN  = 128;
static constexpr int BK  = 32;
static constexpr int LDK = 40;   // padded LDS row stride (bf16), 80 B: 16B aligned

__global__ __launch_bounds__(256)
void gemm_gx(const bf16_t* __restrict__ A,    // [SEQ, IN] bf16
             const bf16_t* __restrict__ B,    // [NO,  IN] bf16 (W_ih)
             const float*  __restrict__ bias, // [NO] (b_ih)
             float* __restrict__ gx)          // [SEQ, NO] f32
{
    __shared__ bf16_t As[BM * LDK];
    __shared__ bf16_t Bs[BN * LDK];

    const int tid   = threadIdx.x;
    const int lane  = tid & 31;
    const int wave  = tid >> 5;
    const int waveM = wave & 3;   // 0..3 -> 4*32 = 128 rows
    const int waveN = wave >> 2;  // 0..1 -> 2*64 = 128 cols
    const int m0 = blockIdx.y * BM;
    const int n0 = blockIdx.x * BN;

    v8f acc[2][4] = {};

    // cooperative tile-load mapping: thread -> (row, 16-elem half)
    const int ldRow  = tid >> 1;        // 0..127
    const int ldHalf = (tid & 1) * 16;  // 0 or 16

    // fragment lane mapping (per ISA wave32 layouts)
    const int am  = lane & 15;          // A: M row within tile
    const int akb = (lane >> 4) * 8;    // A: K sub-block base (0 or 8)
    const int bn  = lane & 15;          // B: N col within tile
    const int bkh = (lane >> 4) * 8;    // B: K sub-block base (0 or 8)

    for (int kk = 0; kk < IN; kk += BK) {
        __syncthreads();   // previous iteration's fragment reads done
        // stage tiles straight into LDS (no VGPR staging, ASYNCcnt-tracked)
        async_copy_32B(A + (size_t)(m0 + ldRow) * IN + kk + ldHalf,
                       &As[ldRow * LDK + ldHalf]);
        async_copy_32B(B + (size_t)(n0 + ldRow) * IN + kk + ldHalf,
                       &Bs[ldRow * LDK + ldHalf]);
        wait_async0();     // own async writes landed in LDS
        __syncthreads();   // everyone's writes landed

        // A fragments: lane holds row m, K = {akb..akb+7, 16+akb..16+akb+7}
        v16bf afrag[2];
#pragma unroll
        for (int mt = 0; mt < 2; ++mt) {
            const int row = waveM * 32 + mt * 16 + am;
            union { v16bf v; v8bf h[2]; } u;
            u.h[0] = *(const v8bf*)&As[row * LDK + akb];
            u.h[1] = *(const v8bf*)&As[row * LDK + 16 + akb];
            afrag[mt] = u.v;
        }
        // B fragments: lane holds col n, K = {bkh..bkh+7, 16+bkh..16+bkh+7}
        v16bf bfrag[4];
#pragma unroll
        for (int nt = 0; nt < 4; ++nt) {
            const int row = waveN * 64 + nt * 16 + bn;
            union { v16bf v; v8bf h[2]; } u;
            u.h[0] = *(const v8bf*)&Bs[row * LDK + bkh];
            u.h[1] = *(const v8bf*)&Bs[row * LDK + 16 + bkh];
            bfrag[nt] = u.v;
        }
#pragma unroll
        for (int mt = 0; mt < 2; ++mt)
#pragma unroll
            for (int nt = 0; nt < 4; ++nt)
                acc[mt][nt] = __builtin_amdgcn_wmma_f32_16x16x32_bf16(
                    false, afrag[mt], false, bfrag[nt],
                    (short)0, acc[mt][nt], false, false);
    }

    // Epilogue: D layout — lane = (m>=8 ? 16:0) + n, VGPR v holds m = 8*(lane>>4)+v
    const int dn   = lane & 15;
    const int dmhi = (lane >> 4) * 8;
#pragma unroll
    for (int mt = 0; mt < 2; ++mt)
#pragma unroll
        for (int nt = 0; nt < 4; ++nt) {
            const int col = n0 + waveN * 64 + nt * 16 + dn;
            const float bv = bias[col];
#pragma unroll
            for (int vr = 0; vr < 8; ++vr) {
                const int row = m0 + waveM * 32 + mt * 16 + dmhi + vr;
                gx[(size_t)row * NO + col] = acc[mt][nt][vr] + bv;
            }
        }
}

// ------------------------ Phase 2: persistent scan --------------------------
// 64 workgroups; WG g owns h-indices [16g, 16g+16) -> 48 rows of W_hh
// (r/z/n gates). Each of 8 waves computes 6 row dot-products per step.
// h double-buffered in global; monotonic atomic counter = grid barrier.
static constexpr int GWG = 64;

__global__ __launch_bounds__(256)
void gru_scan(const float* __restrict__ W_hh,  // [NO, HID]
              const float* __restrict__ b_hh,  // [NO]
              const float* __restrict__ gx,    // [SEQ, NO]
              float* __restrict__ hbuf,        // [2, HID]
              unsigned* __restrict__ bar,      // monotonic arrival counter
              float* __restrict__ out)         // [SEQ, HID]
{
    __shared__ float h_lds[HID];
    __shared__ float ghs[48];

    const int tid  = threadIdx.x;
    const int lane = tid & 31;
    const int wave = tid >> 5;
    const int base = blockIdx.x * 16;

    // cache per-wave row ids and hidden biases
    int   wrow[6];
    float wbias[6];
#pragma unroll
    for (int rr = 0; rr < 6; ++rr) {
        const int r    = wave * 6 + rr;        // 0..47
        const int gate = r >> 4;               // 0=r, 1=z, 2=n
        const int i    = r & 15;
        wrow[rr]  = gate * HID + base + i;
        wbias[rr] = b_hh[wrow[rr]];
    }

    for (int t = 0; t < SEQ; ++t) {
        const float* hin  = hbuf + (size_t)(t & 1) * HID;
        float*       hout = hbuf + (size_t)((t + 1) & 1) * HID;

        // 1) broadcast h into LDS
        *(float4*)&h_lds[tid * 4] = *(const float4*)(hin + tid * 4);
        __syncthreads();

        // 2) gh = W_hh @ h + b_hh  (wave-per-row, float4 loads from L2)
#pragma unroll
        for (int rr = 0; rr < 6; ++rr) {
            const float* wp = W_hh + (size_t)wrow[rr] * HID;
            float s = 0.f;
#pragma unroll
            for (int j = 0; j < 8; ++j) {
                const int k = 4 * lane + 128 * j;
                const float4 w4 = *(const float4*)(wp + k);
                const float4 h4 = *(const float4*)&h_lds[k];
                s += w4.x * h4.x + w4.y * h4.y + w4.z * h4.z + w4.w * h4.w;
            }
#pragma unroll
            for (int off = 16; off > 0; off >>= 1)
                s += __shfl_down(s, off, 32);
            if (lane == 0) ghs[wave * 6 + rr] = s + wbias[rr];
        }
        __syncthreads();

        // 3) fused gate math for the 16 owned hidden units
        if (tid < 16) {
            const int    gi  = base + tid;
            const float* gxt = gx + (size_t)t * NO;
            const float xr = gxt[gi], xz = gxt[HID + gi], xn = gxt[2 * HID + gi];
            const float ghr = ghs[tid], ghz = ghs[16 + tid], ghn = ghs[32 + tid];
            const float r = 1.f / (1.f + __expf(-(xr + ghr)));
            const float z = 1.f / (1.f + __expf(-(xz + ghz)));
            const float n = tanhf(xn + r * ghn);
            const float hn = (1.f - z) * n + z * h_lds[gi];
            hout[gi] = hn;
            out[(size_t)t * HID + gi] = hn;
            __threadfence();
        }
        __syncthreads();

        // 4) device-wide barrier (monotonic counter, s_sleep backoff)
        if (tid == 0) {
            __threadfence();
            atomicAdd(bar, 1u);
            const unsigned target = (unsigned)(GWG * (t + 1));
            while (__hip_atomic_load(bar, __ATOMIC_ACQUIRE,
                                     __HIP_MEMORY_SCOPE_AGENT) < target) {
                __builtin_amdgcn_s_sleep(1);
            }
        }
        __syncthreads();
    }
}

// ------------------------------- launcher -----------------------------------
extern "C" void kernel_launch(void* const* d_in, const int* in_sizes, int n_in,
                              void* d_out, int out_size, void* d_ws, size_t ws_size,
                              hipStream_t stream) {
    const float* inp  = (const float*)d_in[0];
    const float* W_ih = (const float*)d_in[1];
    const float* W_hh = (const float*)d_in[2];
    const float* b_ih = (const float*)d_in[3];
    const float* b_hh = (const float*)d_in[4];
    float* out = (float*)d_out;

    // workspace layout
    char* ws = (char*)d_ws;
    bf16_t* Abf = (bf16_t*)ws;                                  // 16 MB
    size_t off = (size_t)SEQ * IN * sizeof(bf16_t);
    bf16_t* Bbf = (bf16_t*)(ws + off);                          // 6 MB
    off += (size_t)NO * IN * sizeof(bf16_t);
    float* gxw = (float*)(ws + off);                            // 96 MB
    off += (size_t)SEQ * NO * sizeof(float);
    float* hbuf = (float*)(ws + off);                           // 8 KB
    off += (size_t)2 * HID * sizeof(float);
    unsigned* bar = (unsigned*)(ws + off);                      // 4 B

    // reset h0 = 0 and barrier counter = 0 (graph-capture-safe)
    hipMemsetAsync(hbuf, 0, (size_t)2 * HID * sizeof(float) + 64, stream);

    // f32 -> bf16 conversions
    cvt_bf16<<<2048, 256, 0, stream>>>(inp,  Abf, SEQ * IN);
    cvt_bf16<<<2048, 256, 0, stream>>>(W_ih, Bbf, NO * IN);

    // Phase 1: gx = inp @ W_ih^T + b_ih
    dim3 ggrid(NO / BN, SEQ / BM);  // 24 x 64
    gemm_gx<<<ggrid, 256, 0, stream>>>(Abf, Bbf, b_ih, gxw);

    // Phase 2: sequential GRU scan
    gru_scan<<<GWG, 256, 0, stream>>>(W_hh, b_hh, gxw, hbuf, bar, out);
}
